// AST_42288247996593
// MI455X (gfx1250) — matrix operand order repaired and checked by reading, not compiled, not run on hardware
//
#include <hip/hip_runtime.h>
#include <hip/hip_bf16.h>
#include <math.h>

// ---------------------------------------------------------------------------
// ViT-base forward (B=64, N=171 tokens, D=768, 12 layers, 12 heads x 64)
// GEMMs: v_wmma_f32_16x16x32_bf16, 64x256 block tiles, LDS-staged A tiles via
// global_load_async_to_lds_b128 (double buffered, s_wait_asynccnt).
// ---------------------------------------------------------------------------

typedef __attribute__((ext_vector_type(16))) __bf16 v16bf;
typedef __attribute__((ext_vector_type(8)))  float  v8f;
typedef __bf16 bf16_t;

#define D_      768
#define NHEAD   12
#define HD      64
#define NTOK    171
#define BATCH   64
#define NPATCH  170              // 5 x 34 patches (50/10, floor(345/10))
#define MTOT    (BATCH * NTOK)   // 10944 (= 171 * 64)
#define MPAD    (MTOT + 32)      // padding rows for head-window overreads
#define MPATCH  (BATCH * NPATCH) // 10880 (= 170 * 64)
#define KPATCH  608              // 600 padded to multiple of 32
#define FF      3072
#define TD      2304
#define SROWS   176              // 171 padded to 16
#define SCOLS   192              // 171 padded to 32 (K dim of attn@V)

// ---------------------------- WMMA helpers ---------------------------------

__device__ __forceinline__ v8f wmma_bf16(v16bf a, v16bf b, v8f c) {
  return __builtin_amdgcn_wmma_f32_16x16x32_bf16(
      /*neg_a=*/false, a, /*neg_b=*/false, b,
      /*c_mod=*/(short)0, c, /*reuse_a=*/false, /*reuse_b=*/false);
}

// A fragment: rows m0..m0+15, K chunk k0..k0+31, A row-major [*, lda] bf16.
// Lane L holds row m0 + L%16; element j -> k0 + hi*16 + j.
__device__ __forceinline__ v16bf load_a_frag(const bf16_t* __restrict__ A,
                                             int lda, int m0, int k0) {
  int lane = threadIdx.x & 31;
  const bf16_t* p = A + (size_t)(m0 + (lane & 15)) * lda + k0 + ((lane >> 4) * 16);
  v16bf r;
#pragma unroll
  for (int j = 0; j < 16; ++j) r[j] = p[j];
  return r;
}

// B fragment from TRANSPOSED weights Bt[N, ldb] (Bt[n,k] = W[k,n]).
__device__ __forceinline__ v16bf load_bt_frag(const bf16_t* __restrict__ Bt,
                                              int ldb, int n0, int k0) {
  int lane = threadIdx.x & 31;
  const bf16_t* p = Bt + (size_t)(n0 + (lane & 15)) * ldb + k0 + ((lane >> 4) * 16);
  v16bf r;
#pragma unroll
  for (int j = 0; j < 16; ++j) r[j] = p[j];
  return r;
}

// B fragment from NON-transposed B[K, ldb] (B[k,n]); strided per-lane loads.
__device__ __forceinline__ v16bf load_bn_frag(const bf16_t* __restrict__ Bm,
                                              int ldb, int k0, int n0) {
  int lane = threadIdx.x & 31;
  const bf16_t* p = Bm + (size_t)(k0 + ((lane >> 4) * 16)) * ldb + (n0 + (lane & 15));
  v16bf r;
#pragma unroll
  for (int j = 0; j < 16; ++j) r[j] = p[(size_t)j * ldb];
  return r;
}

// ------------------ 64x256 block GEMM core (LDS-staged A) -------------------
// Block = 256 threads = 8 waves arranged 2(m) x 4(n). Each wave owns a 32x64
// output tile -> v8f acc[2][4]. A tile (64 rows x 32 k) is async-copied into
// LDS (double buffered); B fragments stream from global (bf16 [N,K]).
__device__ __forceinline__ void gemm_block_64x256(
    const bf16_t* __restrict__ A, int lda, const bf16_t* __restrict__ Wt,
    int K, int m0, int n0, v8f (&acc)[2][4]) {
  __shared__ bf16_t shA[2][64][32];          // 8 KB
  const int tid  = threadIdx.x;
  const int lane = tid & 31;
  const int wave = tid >> 5;
  const int mw = wave & 1, nw = wave >> 1;
  const int nsteps = K >> 5;

  const int srow = tid >> 2;                 // 0..63
  const int sseg = (tid & 3) * 8;            // 0,8,16,24 (16B chunks)

  // async stage: 256 threads x 16B = full 64x32 bf16 tile
  auto stage = [&](int buf, int k0) {
    unsigned lds = (unsigned)(size_t)&shA[buf][srow][sseg];
    unsigned long long ga =
        (unsigned long long)(size_t)(A + (size_t)(m0 + srow) * lda + k0 + sseg);
    asm volatile("global_load_async_to_lds_b128 %0, %1, off"
                 :: "v"(lds), "v"(ga) : "memory");
  };

  stage(0, 0);
  for (int ks = 0; ks < nsteps; ++ks) {
    const int buf = ks & 1;
    const int k0  = ks << 5;
    if (ks + 1 < nsteps) {
      stage(buf ^ 1, k0 + 32);
      asm volatile("s_wait_asynccnt 0x1" ::: "memory");  // current buf landed
    } else {
      asm volatile("s_wait_asynccnt 0x0" ::: "memory");
    }
    __syncthreads();

    const bf16_t* As = &shA[buf][mw * 32][0];
    v16bf a0 = load_a_frag(As, 32, 0, 0);
    v16bf a1 = load_a_frag(As, 32, 16, 0);
#pragma unroll
    for (int j = 0; j < 4; ++j) {
      const int nj = n0 + nw * 64 + j * 16;
      __builtin_prefetch(Wt + (size_t)(nj + (lane & 15)) * K + k0 + 64, 0, 1);
      v16bf b = load_bt_frag(Wt, K, nj, k0);
      acc[0][j] = wmma_bf16(a0, b, acc[0][j]);
      acc[1][j] = wmma_bf16(a1, b, acc[1][j]);
    }
    __syncthreads();                         // readers done before re-stage
  }
}

// ---------------------------- prep kernels ---------------------------------

// im2col: x[64,6,50,345] -> Apatch bf16 [10880, 608] (K 600..607 zero)
__global__ void k_im2col(const float* __restrict__ x, bf16_t* __restrict__ ap) {
  int idx = blockIdx.x * blockDim.x + threadIdx.x;
  if (idx >= MPATCH * KPATCH) return;
  int m = idx / KPATCH, k = idx - m * KPATCH;
  float v = 0.f;
  if (k < 600) {
    int b  = m / NPATCH, p = m - b * NPATCH;
    int ph = p / 34,     pw = p - ph * 34;
    int c  = k / 100,    rem = k - c * 100;
    int i  = rem / 10,   j = rem - i * 10;
    v = x[(((size_t)b * 6 + c) * 50 + (ph * 10 + i)) * 345 + (pw * 10 + j)];
  }
  ap[idx] = (bf16_t)v;
}

// conv_w [768,600] -> bf16 [768,608]
__global__ void k_convw(const float* __restrict__ w, bf16_t* __restrict__ wt) {
  int idx = blockIdx.x * blockDim.x + threadIdx.x;
  if (idx >= D_ * KPATCH) return;
  int n = idx / KPATCH, k = idx - n * KPATCH;
  wt[idx] = (bf16_t)((k < 600) ? w[(size_t)n * 600 + k] : 0.f);
}

// Generic transpose+convert: W f32 [K, N] -> Wt bf16 [N, K]
__global__ void k_wt(const float* __restrict__ w, bf16_t* __restrict__ wt,
                     int K, int N) {
  int idx = blockIdx.x * blockDim.x + threadIdx.x;
  if (idx >= K * N) return;
  int n = idx / K, k = idx - n * K;
  wt[idx] = (bf16_t)w[(size_t)k * N + n];
}

// h[b,0,:] = cls + pos[0,:]
__global__ void k_cls(const float* __restrict__ cls, const float* __restrict__ pos,
                      float* __restrict__ h) {
  int idx = blockIdx.x * blockDim.x + threadIdx.x;
  if (idx >= BATCH * D_) return;
  int b = idx / D_, n = idx - b * D_;
  h[(size_t)b * NTOK * D_ + n] = cls[n] + pos[n];
}

// ---------------------------- GEMM kernels ---------------------------------

// Patch-embed GEMM: Apatch[10880,608] @ Wconv^T -> h[b, 1+p, :] (+bias+pos)
__global__ void k_patch_gemm(const bf16_t* __restrict__ ap, const bf16_t* __restrict__ wt,
                             const float* __restrict__ cb, const float* __restrict__ pos,
                             float* __restrict__ h) {
  const int m0 = blockIdx.x * 64, n0 = blockIdx.y * 256;
  v8f acc[2][4] = {};
  gemm_block_64x256(ap, KPATCH, wt, KPATCH, m0, n0, acc);
  const int lane = threadIdx.x & 31, wave = threadIdx.x >> 5;
  const int mw = wave & 1, nw = wave >> 1;
  const int mhi = (lane >> 4) * 8;
#pragma unroll
  for (int i = 0; i < 2; ++i)
#pragma unroll
    for (int j = 0; j < 4; ++j) {
      int gn = n0 + nw * 64 + j * 16 + (lane & 15);
      float bv = cb[gn];
#pragma unroll
      for (int r = 0; r < 8; ++r) {
        int m = m0 + mw * 32 + i * 16 + mhi + r;
        int b = m / NPATCH, p = m - b * NPATCH;
        h[((size_t)b * NTOK + 1 + p) * D_ + gn] =
            acc[i][j][r] + bv + pos[(size_t)(1 + p) * D_ + gn];
      }
    }
}

// Generic GEMM: A bf16 [MTOT, lda] @ Wt bf16 [N, K] -> bf16 out (+bias, opt GELU)
__global__ void k_gemm_bf16out(const bf16_t* __restrict__ A, int lda,
                               const bf16_t* __restrict__ Wt, const float* __restrict__ bias,
                               bf16_t* __restrict__ out, int ldo, int K, int act) {
  const int m0 = blockIdx.x * 64, n0 = blockIdx.y * 256;
  v8f acc[2][4] = {};
  gemm_block_64x256(A, lda, Wt, K, m0, n0, acc);
  const int lane = threadIdx.x & 31, wave = threadIdx.x >> 5;
  const int mw = wave & 1, nw = wave >> 1;
  const int mhi = (lane >> 4) * 8;
#pragma unroll
  for (int i = 0; i < 2; ++i)
#pragma unroll
    for (int j = 0; j < 4; ++j) {
      int gn = n0 + nw * 64 + j * 16 + (lane & 15);
      float bv = bias[gn];
#pragma unroll
      for (int r = 0; r < 8; ++r) {
        int m = m0 + mw * 32 + i * 16 + mhi + r;
        float c = acc[i][j][r] + bv;
        if (act) c = 0.5f * c * (1.f + erff(c * 0.70710678118654752f));
        out[(size_t)m * ldo + gn] = (bf16_t)c;
      }
    }
}

// Generic GEMM with residual: h[m, n] += A@Wt + bias   (N = 768)
__global__ void k_gemm_residual(const bf16_t* __restrict__ A, int lda,
                                const bf16_t* __restrict__ Wt, const float* __restrict__ bias,
                                float* __restrict__ h, int K) {
  const int m0 = blockIdx.x * 64, n0 = blockIdx.y * 256;
  v8f acc[2][4] = {};
  gemm_block_64x256(A, lda, Wt, K, m0, n0, acc);
  const int lane = threadIdx.x & 31, wave = threadIdx.x >> 5;
  const int mw = wave & 1, nw = wave >> 1;
  const int mhi = (lane >> 4) * 8;
#pragma unroll
  for (int i = 0; i < 2; ++i)
#pragma unroll
    for (int j = 0; j < 4; ++j) {
      int gn = n0 + nw * 64 + j * 16 + (lane & 15);
      float bv = bias[gn];
#pragma unroll
      for (int r = 0; r < 8; ++r) {
        int m = m0 + mw * 32 + i * 16 + mhi + r;
        h[(size_t)m * D_ + gn] += acc[i][j][r] + bv;
      }
    }
}

// scores[bh, t, j] = q[t,:] . k[j,:]    (per batch*head; K = 64)
__global__ void k_scores(const bf16_t* __restrict__ qkv, float* __restrict__ sc) {
  int bh = blockIdx.x;
  int b = bh / NHEAD, hh = bh - b * NHEAD;
  int wave = threadIdx.x >> 5;
  int nt = blockIdx.z * 8 + wave;
  if (nt >= 11) return;                       // wave-uniform, EXEC stays full
  int mt = blockIdx.y;
  const bf16_t* qb = qkv + (size_t)(b * NTOK) * TD + hh * HD;   // q block
  const bf16_t* kb = qb + D_;                                    // k block
  v8f acc = {};
  acc = wmma_bf16(load_a_frag(qb, TD, mt * 16, 0),  load_bt_frag(kb, TD, nt * 16, 0),  acc);
  acc = wmma_bf16(load_a_frag(qb, TD, mt * 16, 32), load_bt_frag(kb, TD, nt * 16, 32), acc);
  int lane = threadIdx.x & 31;
  int gn  = nt * 16 + (lane & 15);
  int mhi = (lane >> 4) * 8;
  float* srow = sc + (size_t)bh * SROWS * SCOLS;
#pragma unroll
  for (int r = 0; r < 8; ++r)
    srow[(size_t)(mt * 16 + mhi + r) * SCOLS + gn] = acc[r];
}

// softmax over j<171 with scale 1/8; write bf16 probs, zero pad to 192 cols
__global__ void k_softmax(const float* __restrict__ sc, bf16_t* __restrict__ pr) {
  int row = blockIdx.x;                 // 0 .. 768*171-1
  int bh = row / NTOK, t = row - bh * NTOK;
  const float* s = sc + ((size_t)bh * SROWS + t) * SCOLS;
  bf16_t*      p = pr + ((size_t)bh * SROWS + t) * SCOLS;
  int j = threadIdx.x;                  // 256 threads
  float v = (j < NTOK) ? s[j] * 0.125f : -3.0e38f;
  __shared__ float red[256];
  red[j] = v; __syncthreads();
  for (int o = 128; o > 0; o >>= 1) {
    if (j < o) red[j] = fmaxf(red[j], red[j + o]);
    __syncthreads();
  }
  float mx = red[0]; __syncthreads();
  float e = (j < NTOK) ? expf(v - mx) : 0.f;
  red[j] = e; __syncthreads();
  for (int o = 128; o > 0; o >>= 1) {
    if (j < o) red[j] += red[j + o];
    __syncthreads();
  }
  float inv = 1.f / red[0];
  if (j < SCOLS) p[j] = (bf16_t)(e * inv);
}

// o[t, head*64+d] = probs[t,:] @ v[:, d]   (K = 192, zero-padded cols)
__global__ void k_attn_v(const bf16_t* __restrict__ pr, const bf16_t* __restrict__ qkv,
                         bf16_t* __restrict__ o) {
  int bh = blockIdx.x;
  int b = bh / NHEAD, hh = bh - b * NHEAD;
  int mt = blockIdx.y;
  int wave = threadIdx.x >> 5;          // blockDim 128 -> 4 waves -> N=64
  int n0 = wave * 16;
  const bf16_t* A = pr + (size_t)bh * SROWS * SCOLS;
  const bf16_t* V = qkv + (size_t)(b * NTOK) * TD + 2 * D_ + hh * HD;
  v8f acc = {};
  for (int k0 = 0; k0 < SCOLS; k0 += 32)
    acc = wmma_bf16(load_a_frag(A, SCOLS, mt * 16, k0), load_bn_frag(V, TD, k0, n0), acc);
  int lane = threadIdx.x & 31;
  int gn  = hh * HD + n0 + (lane & 15);
  int mhi = (lane >> 4) * 8;
#pragma unroll
  for (int r = 0; r < 8; ++r) {
    int t = mt * 16 + mhi + r;
    if (t < NTOK) o[(size_t)(b * NTOK + t) * D_ + gn] = (bf16_t)acc[r];
  }
}

// ---------------------------- LN / head ------------------------------------

__global__ void k_ln(const float* __restrict__ h, const float* __restrict__ w,
                     const float* __restrict__ bb, bf16_t* __restrict__ y) {
  int row = blockIdx.x, tid = threadIdx.x;
  const float* x = h + (size_t)row * D_;
  float v0 = x[tid], v1 = x[tid + 256], v2 = x[tid + 512];
  __shared__ float ss[256], sq[256];
  ss[tid] = v0 + v1 + v2;
  sq[tid] = v0 * v0 + v1 * v1 + v2 * v2;
  __syncthreads();
  for (int o = 128; o > 0; o >>= 1) {
    if (tid < o) { ss[tid] += ss[tid + o]; sq[tid] += sq[tid + o]; }
    __syncthreads();
  }
  float mean = ss[0] * (1.f / 768.f);
  float var  = sq[0] * (1.f / 768.f) - mean * mean;
  float rs = rsqrtf(var + 1e-5f);
  bf16_t* yr = y + (size_t)row * D_;
  yr[tid]       = (bf16_t)((v0 - mean) * rs * w[tid]       + bb[tid]);
  yr[tid + 256] = (bf16_t)((v1 - mean) * rs * w[tid + 256] + bb[tid + 256]);
  yr[tid + 512] = (bf16_t)((v2 - mean) * rs * w[tid + 512] + bb[tid + 512]);
}

__global__ void k_ln_cls(const float* __restrict__ h, const float* __restrict__ w,
                         const float* __restrict__ bb, float* __restrict__ out) {
  int b = blockIdx.x, tid = threadIdx.x;
  const float* x = h + (size_t)b * NTOK * D_;     // CLS row
  float v0 = x[tid], v1 = x[tid + 256], v2 = x[tid + 512];
  __shared__ float ss[256], sq[256];
  ss[tid] = v0 + v1 + v2;
  sq[tid] = v0 * v0 + v1 * v1 + v2 * v2;
  __syncthreads();
  for (int o = 128; o > 0; o >>= 1) {
    if (tid < o) { ss[tid] += ss[tid + o]; sq[tid] += sq[tid + o]; }
    __syncthreads();
  }
  float mean = ss[0] * (1.f / 768.f);
  float var  = sq[0] * (1.f / 768.f) - mean * mean;
  float rs = rsqrtf(var + 1e-5f);
  float* orow = out + (size_t)b * D_;
  orow[tid]       = (v0 - mean) * rs * w[tid]       + bb[tid];
  orow[tid + 256] = (v1 - mean) * rs * w[tid + 256] + bb[tid + 256];
  orow[tid + 512] = (v2 - mean) * rs * w[tid + 512] + bb[tid + 512];
}

__global__ void k_head(const float* __restrict__ cls, const float* __restrict__ hw,
                       const float* __restrict__ hb, float* __restrict__ out) {
  int idx = blockIdx.x * blockDim.x + threadIdx.x;
  if (idx >= BATCH * 3) return;
  int b = idx / 3, c = idx - b * 3;
  float s = hb[c];
  for (int n = 0; n < D_; ++n) s += cls[(size_t)b * D_ + n] * hw[n * 3 + c];
  out[idx] = s;
}

// ---------------------------- launcher -------------------------------------

extern "C" void kernel_launch(void* const* d_in, const int* in_sizes, int n_in,
                              void* d_out, int out_size, void* d_ws, size_t ws_size,
                              hipStream_t stream) {
  (void)in_sizes; (void)n_in; (void)out_size; (void)ws_size;
  const float* x        = (const float*)d_in[0];
  const float* conv_w   = (const float*)d_in[1];
  const float* conv_b   = (const float*)d_in[2];
  const float* cls_tok  = (const float*)d_in[3];
  const float* pos      = (const float*)d_in[4];
  const float* ln1_w    = (const float*)d_in[5];
  const float* ln1_b    = (const float*)d_in[6];
  const float* qkv_w    = (const float*)d_in[7];
  const float* qkv_b    = (const float*)d_in[8];
  const float* proj_w   = (const float*)d_in[9];
  const float* proj_b   = (const float*)d_in[10];
  const float* ln2_w    = (const float*)d_in[11];
  const float* ln2_b    = (const float*)d_in[12];
  const float* fc1_w    = (const float*)d_in[13];
  const float* fc1_b    = (const float*)d_in[14];
  const float* fc2_w    = (const float*)d_in[15];
  const float* fc2_b    = (const float*)d_in[16];
  const float* norm_w   = (const float*)d_in[17];
  const float* norm_b   = (const float*)d_in[18];
  const float* head_w   = (const float*)d_in[19];
  const float* head_b   = (const float*)d_in[20];

  char* base = (char*)d_ws;
  size_t off = 0;
  auto alloc = [&](size_t bytes) -> char* {
    char* p = base + off;
    off = (off + bytes + 255) & ~(size_t)255;
    return p;
  };
  float*  h    = (float*) alloc((size_t)MTOT  * D_ * 4);
  bf16_t* y    = (bf16_t*)alloc((size_t)MTOT  * D_ * 2);
  bf16_t* qkv  = (bf16_t*)alloc((size_t)MPAD  * TD * 2);
  float*  sc   = (float*) alloc((size_t)BATCH * NHEAD * SROWS * SCOLS * 4);
  bf16_t* pr   = (bf16_t*)alloc((size_t)BATCH * NHEAD * SROWS * SCOLS * 2);
  bf16_t* ob   = (bf16_t*)alloc((size_t)MTOT  * D_ * 2);
  bf16_t* g    = (bf16_t*)alloc((size_t)MTOT  * FF * 2);
  bf16_t* ap   = (bf16_t*)alloc((size_t)MPATCH * KPATCH * 2);
  bf16_t* wtc  = (bf16_t*)alloc((size_t)D_ * KPATCH * 2);
  bf16_t* wtq  = (bf16_t*)alloc((size_t)TD * D_ * 2);
  bf16_t* wtp  = (bf16_t*)alloc((size_t)D_ * D_ * 2);
  bf16_t* wt1  = (bf16_t*)alloc((size_t)FF * D_ * 2);
  bf16_t* wt2  = (bf16_t*)alloc((size_t)D_ * FF * 2);
  float*  clsb = (float*) alloc((size_t)BATCH * D_ * 4);

  // Zero pads so no garbage flows through WMMA K-dims (capture-safe memsets).
  hipMemsetAsync(qkv + (size_t)MTOT * TD, 0, (size_t)(MPAD - MTOT) * TD * 2, stream);
  hipMemsetAsync(pr, 0, (size_t)BATCH * NHEAD * SROWS * SCOLS * 2, stream);

  // Patch embed
  k_im2col<<<(MPATCH * KPATCH + 255) / 256, 256, 0, stream>>>(x, ap);
  k_convw <<<(D_ * KPATCH + 255) / 256, 256, 0, stream>>>(conv_w, wtc);
  k_patch_gemm<<<dim3(MPATCH / 64, D_ / 256), 256, 0, stream>>>(ap, wtc, conv_b, pos, h);
  k_cls<<<(BATCH * D_ + 255) / 256, 256, 0, stream>>>(cls_tok, pos, h);

  for (int i = 0; i < 12; ++i) {
    // Per-layer weight transpose + bf16 convert
    k_wt<<<(D_ * TD + 255) / 256, 256, 0, stream>>>(qkv_w  + (size_t)i * D_ * TD, wtq, D_, TD);
    k_wt<<<(D_ * D_ + 255) / 256, 256, 0, stream>>>(proj_w + (size_t)i * D_ * D_, wtp, D_, D_);
    k_wt<<<(D_ * FF + 255) / 256, 256, 0, stream>>>(fc1_w  + (size_t)i * D_ * FF, wt1, D_, FF);
    k_wt<<<(FF * D_ + 255) / 256, 256, 0, stream>>>(fc2_w  + (size_t)i * FF * D_, wt2, FF, D_);

    // Attention block
    k_ln<<<MTOT, 256, 0, stream>>>(h, ln1_w + i * D_, ln1_b + i * D_, y);
    k_gemm_bf16out<<<dim3(MTOT / 64, TD / 256), 256, 0, stream>>>(
        y, D_, wtq, qkv_b + i * TD, qkv, TD, D_, 0);
    k_scores<<<dim3(BATCH * NHEAD, SROWS / 16, 2), 256, 0, stream>>>(qkv, sc);
    k_softmax<<<BATCH * NHEAD * NTOK, 256, 0, stream>>>(sc, pr);
    k_attn_v<<<dim3(BATCH * NHEAD, SROWS / 16), 128, 0, stream>>>(pr, qkv, ob);
    k_gemm_residual<<<dim3(MTOT / 64, D_ / 256), 256, 0, stream>>>(
        ob, D_, wtp, proj_b + i * D_, h, D_);

    // MLP block
    k_ln<<<MTOT, 256, 0, stream>>>(h, ln2_w + i * D_, ln2_b + i * D_, y);
    k_gemm_bf16out<<<dim3(MTOT / 64, FF / 256), 256, 0, stream>>>(
        y, D_, wt1, fc1_b + i * FF, g, FF, D_, 1);
    k_gemm_residual<<<dim3(MTOT / 64, D_ / 256), 256, 0, stream>>>(
        g, FF, wt2, fc2_b + i * D_, h, FF);
  }

  k_ln_cls<<<BATCH, 256, 0, stream>>>(h, norm_w, norm_b, clsb);
  k_head<<<1, 192, 0, stream>>>(clsb, head_w, head_b, (float*)d_out);
}
